// BLSTM_30468497998006
// MI455X (gfx1250) — matrix operand order repaired and checked by reading, not compiled, not run on hardware
//
#include <hip/hip_runtime.h>

// ============================================================================
// Bidirectional LSTM for MI455X (gfx1250), wave32 + WMMA bf16 + async-to-LDS.
//
// phase0: init barriers + bf16 h double-buffers
// phase1: xz = x@Wk + b  (M=16384, N=2048pad, K=512) via v_wmma_f32_16x16x32_bf16
// phase2: persistent recurrence, Wr slice resident in LDS, h staged per step
//         with GLOBAL_LOAD_ASYNC_TO_LDS_B128 (ASYNCcnt), per-step agent barrier.
// ============================================================================

typedef __bf16   bf16;
typedef bf16     v16bf __attribute__((ext_vector_type(16)));
typedef float    v8f   __attribute__((ext_vector_type(8)));
typedef float    f4    __attribute__((ext_vector_type(4)));
typedef unsigned u4    __attribute__((ext_vector_type(4)));

static constexpr int B_     = 64;
static constexpr int T_     = 256;
static constexpr int D_     = 512;   // input feature dim (K of phase1)
static constexpr int U_     = 500;   // units
static constexpr int UPAD   = 512;   // units padded (K of phase2)
static constexpr int NCOL   = 2000;  // 4*U
static constexpr int NPAD   = 2048;  // padded gate columns
static constexpr int ROWS   = B_ * T_;      // 16384
static constexpr int WG_DIR = 16;           // workgroups per direction (phase2)
static constexpr int HSTR   = 520;          // LDS h row stride (halfs), 1040B = 65*16 (16B aligned, bank-safe)
static constexpr int WSTR   = 520;          // LDS WrT col stride (halfs)

__device__ __forceinline__ float sigm(float x) { return 1.0f / (1.0f + __expf(-x)); }

__device__ __forceinline__ void ldsm_a(v16bf& frag, const bf16* p) {
    // 16-bit A 16x32 frag: lane<16 holds K0..15; lane>=16 holds K16..31
    f4* d = reinterpret_cast<f4*>(&frag);
    d[0] = *reinterpret_cast<const f4*>(p);
    d[1] = *reinterpret_cast<const f4*>(p + 8);
}
__device__ __forceinline__ void ldsm_b(v16bf& frag, const bf16* p) {
    // 16-bit B 32x16 frag: lane n<16 holds col n K0..15; lane n+16 holds col n K16..31
    f4* d = reinterpret_cast<f4*>(&frag);
    d[0] = *reinterpret_cast<const f4*>(p);
    d[1] = *reinterpret_cast<const f4*>(p + 8);
}

// ---------------------------------------------------------------------------
// Phase 0: zero barriers, seed h double-buffers (bf16, padded 500->512).
// hbuf layout: [dir][buf(2)][64][512] bf16
// ---------------------------------------------------------------------------
__global__ void lstm_init(const float* __restrict__ h_f, const float* __restrict__ h_b,
                          bf16* __restrict__ hbuf, unsigned* __restrict__ bar) {
    int i = blockIdx.x * blockDim.x + threadIdx.x;
    if (i == 0) { bar[0] = 0u; bar[1] = 0u; }
    int dir = i >> 16;
    int rem = i & 0xFFFF;
    int buf = rem >> 15;
    int r2  = rem & 0x7FFF;
    int b   = r2 >> 9;
    int u   = r2 & 511;
    const float* h0 = dir ? h_b : h_f;
    float v = (buf == 0 && u < U_) ? h0[b * U_ + u] : 0.0f;
    hbuf[i] = (bf16)v;
}

// ---------------------------------------------------------------------------
// Phase 1: xz[dir] = x @ Wk[dir] + b[dir]    (M=16384, N=2000->2048, K=512)
// Block tile 128x128, 8 waves (4 m-quadrants x 2 n-halves), each wave 2x4
// 16x16 tiles, K staged in LDS as bf16 (A row-major, B transposed), stride 40
// halfs to avoid bank conflicts on the b128 fragment reads.
// ---------------------------------------------------------------------------
__global__ void __launch_bounds__(256)
lstm_input_gemm(const float* __restrict__ x,
                const float* __restrict__ Wk_f, const float* __restrict__ Wk_b,
                const float* __restrict__ b_f,  const float* __restrict__ b_b,
                float* __restrict__ xz) {
    __shared__ bf16 As[128 * 40];
    __shared__ bf16 Bs[128 * 40];

    const int dir  = blockIdx.z;
    const float* Wk   = dir ? Wk_b : Wk_f;
    const float* bias = dir ? b_b  : b_f;
    float* out = xz + (size_t)dir * ROWS * NPAD;

    const int tid  = threadIdx.x;
    const int lane = tid & 31;
    const int wave = tid >> 5;
    const int wm   = wave & 3;   // 32-row quadrant
    const int wn   = wave >> 2;  // 64-col half
    const int n    = lane & 15;
    const int hi   = lane >> 4;

    const int row0 = blockIdx.y * 128;
    const int col0 = blockIdx.x * 128;

    v8f acc[2][4];
    for (int mt = 0; mt < 2; ++mt)
        for (int nt = 0; nt < 4; ++nt)
            for (int v = 0; v < 8; ++v) acc[mt][nt][v] = 0.0f;

    for (int kb = 0; kb < D_ / 32; ++kb) {
        // stage A: lanes sweep K (coalesced), rows per iteration
        #pragma unroll
        for (int i = 0; i < 16; ++i) {
            int idx = i * 256 + tid;
            int kk  = idx & 31;
            int r   = idx >> 5;
            As[r * 40 + kk] = (bf16)x[(size_t)(row0 + r) * D_ + kb * 32 + kk];
        }
        // stage B transposed: lanes sweep columns (coalesced)
        #pragma unroll
        for (int i = 0; i < 16; ++i) {
            int idx = i * 256 + tid;
            int c   = idx & 127;
            int kk  = idx >> 7;
            int col = col0 + c;
            float v = (col < NCOL) ? Wk[(size_t)(kb * 32 + kk) * NCOL + col] : 0.0f;
            Bs[c * 40 + kk] = (bf16)v;
        }
        if (kb + 1 < D_ / 32)
            __builtin_prefetch(x + (size_t)row0 * D_ + (kb + 1) * 32, 0, 1);
        __syncthreads();

        v16bf afr[2];
        #pragma unroll
        for (int mt = 0; mt < 2; ++mt)
            ldsm_a(afr[mt], &As[(wm * 32 + mt * 16 + n) * 40 + hi * 16]);
        #pragma unroll
        for (int nt = 0; nt < 4; ++nt) {
            v16bf bfr;
            ldsm_b(bfr, &Bs[(wn * 64 + nt * 16 + n) * 40 + hi * 16]);
            #pragma unroll
            for (int mt = 0; mt < 2; ++mt)
                acc[mt][nt] = __builtin_amdgcn_wmma_f32_16x16x32_bf16(
                    false, afr[mt], false, bfr, (short)0, acc[mt][nt], false, false);
        }
        __syncthreads();
    }

    // epilogue: D-layout (m = v + 8*hi, n = lane%16), add bias, store f32
    #pragma unroll
    for (int nt = 0; nt < 4; ++nt) {
        int cg = col0 + wn * 64 + nt * 16 + n;
        float bv = (cg < NCOL) ? bias[cg] : 0.0f;
        #pragma unroll
        for (int mt = 0; mt < 2; ++mt) {
            #pragma unroll
            for (int v = 0; v < 8; ++v) {
                int rg = row0 + wm * 32 + mt * 16 + v + 8 * hi;
                out[(size_t)rg * NPAD + cg] = acc[mt][nt][v] + bv;
            }
        }
    }
}

// ---------------------------------------------------------------------------
// Phase 2: persistent recurrence. 32 blocks = 2 dirs x 16 WGs; WG owns units
// [32*wg, 32*wg+32) across all 4 gates. WrT slice (128 cols x 512 K, bf16)
// lives in LDS for the whole kernel (~133 KB); h staged per step (~67 KB)
// via GLOBAL_LOAD_ASYNC_TO_LDS_B128 (ASYNCcnt path — no VGPR round trip).
// Wave w: ut = w>>2 (unit subtile), mt = w&3 (batch subtile); cell state c
// stays in VGPRs. Per-step device barrier: release/acquire agent atomics.
// ---------------------------------------------------------------------------
__global__ void __launch_bounds__(256)
lstm_recurrent(const float* __restrict__ Wr_f, const float* __restrict__ Wr_b,
               const float* __restrict__ c_f,  const float* __restrict__ c_b,
               const float* __restrict__ xz,   bf16* __restrict__ hbuf,
               unsigned* __restrict__ bar,     float* __restrict__ y) {
    extern __shared__ char smem[];
    bf16* wrT = reinterpret_cast<bf16*>(smem);                     // [128][WSTR]
    bf16* hsh = reinterpret_cast<bf16*>(smem + 128 * WSTR * 2);    // [64][HSTR]

    const int dir  = blockIdx.x / WG_DIR;
    const int wg   = blockIdx.x % WG_DIR;
    const int tid  = threadIdx.x;
    const int lane = tid & 31;
    const int wave = tid >> 5;
    const int ut   = wave >> 2;  // 0..1 unit subtile
    const int mt   = wave & 3;   // 0..3 batch subtile
    const int n    = lane & 15;
    const int hi   = lane >> 4;

    const float* Wr  = dir ? Wr_b : Wr_f;
    const float* c0p = dir ? c_b  : c_f;
    const float* xzd = xz + (size_t)dir * ROWS * NPAD;
    const int u0     = wg * 32;
    const int uG     = u0 + ut * 16 + n;          // this thread's unit column

    // --- load Wr slice transposed into LDS (once). cl = gate*32 + u_local ---
    for (int idx = tid; idx < 128 * UPAD; idx += blockDim.x) {
        int cl = idx & 127;
        int k  = idx >> 7;
        int g  = cl >> 5;
        int ul = cl & 31;
        int u  = u0 + ul;
        float v = (k < U_ && u < U_) ? Wr[(size_t)k * NCOL + g * U_ + u] : 0.0f;
        wrT[cl * WSTR + k] = (bf16)v;
    }

    // --- cell state in registers (D-layout thread mapping) ---
    float cst[8];
    #pragma unroll
    for (int v = 0; v < 8; ++v) {
        int b = mt * 16 + v + 8 * hi;
        cst[v] = (uG < U_) ? c0p[b * U_ + uG] : 0.0f;
    }

    for (int s = 0; s < T_; ++s) {
        const int t = dir ? (T_ - 1 - s) : s;
        const bf16* hread  = hbuf + ((size_t)dir * 2 + (s & 1))       * (B_ * UPAD);
        bf16*       hwrite = hbuf + ((size_t)dir * 2 + ((s + 1) & 1)) * (B_ * UPAD);

        // stage h[64][512] bf16 -> LDS asynchronously (16B per lane per issue).
        // LDS dest offset = low 32 bits of the generic shared pointer (LDS
        // aperture truncation per ISA 10.2); both sides 16B aligned.
        #pragma unroll
        for (int i = 0; i < (B_ * UPAD) / 8 / 256; ++i) {
            int idx = i * 256 + tid;
            int row = idx >> 6;
            int k8  = idx & 63;
            unsigned           ldsoff = (unsigned)(uintptr_t)(hsh + row * HSTR + k8 * 8);
            unsigned long long gaddr  = (unsigned long long)(uintptr_t)(hread + row * UPAD + k8 * 8);
            asm volatile("global_load_async_to_lds_b128 %0, %1, off"
                         :: "v"(ldsoff), "v"(gaddr) : "memory");
        }

        // prefetch next step's xz seeds while the async copy is in flight
        if (s + 1 < T_) {
            int tn = dir ? (T_ - 2 - s) : (s + 1);
            int b0 = mt * 16 + 8 * hi;
            __builtin_prefetch(xzd + (size_t)(b0 * T_ + tn) * NPAD + uG, 0, 1);
        }

        // accumulators seeded with xz (z = xz_t + h @ Wr)
        v8f acc[4];
        #pragma unroll
        for (int g = 0; g < 4; ++g) {
            #pragma unroll
            for (int v = 0; v < 8; ++v) {
                int b = mt * 16 + v + 8 * hi;
                acc[g][v] = xzd[(size_t)(b * T_ + t) * NPAD + g * U_ + uG];
            }
        }

        asm volatile("s_wait_asynccnt 0" ::: "memory");
        __syncthreads();

        // K loop: 16 x (A frag reused across 4 gate tiles)
        #pragma unroll 4
        for (int kb = 0; kb < UPAD / 32; ++kb) {
            v16bf afr;
            ldsm_a(afr, &hsh[(mt * 16 + n) * HSTR + kb * 32 + hi * 16]);
            #pragma unroll
            for (int g = 0; g < 4; ++g) {
                v16bf bfr;
                ldsm_b(bfr, &wrT[(g * 32 + ut * 16 + n) * WSTR + kb * 32 + hi * 16]);
                acc[g] = __builtin_amdgcn_wmma_f32_16x16x32_bf16(
                    false, afr, false, bfr, (short)0, acc[g], false, false);
            }
        }

        // gates + state update + stores
        #pragma unroll
        for (int v = 0; v < 8; ++v) {
            float iv = sigm(acc[0][v]);
            float fv = sigm(acc[1][v]);
            float gv = tanhf(acc[2][v]);
            float ov = sigm(acc[3][v]);
            float cn = fv * cst[v] + iv * gv;
            cst[v]   = cn;
            float hv = ov * tanhf(cn);
            int b = mt * 16 + v + 8 * hi;
            bool ok = (uG < U_);
            hwrite[b * UPAD + u0 + ut * 16 + n] = (bf16)(ok ? hv : 0.0f);
            if (ok) y[(size_t)(b * T_ + t) * (2 * U_) + dir * U_ + uG] = hv;
        }

        // device-scope split barrier across this direction's 16 WGs
        __syncthreads();
        if (tid == 0) {
            __hip_atomic_fetch_add(&bar[dir], 1u, __ATOMIC_RELEASE, __HIP_MEMORY_SCOPE_AGENT);
            const unsigned target = (unsigned)(s + 1) * WG_DIR;
            while (__hip_atomic_load(&bar[dir], __ATOMIC_ACQUIRE, __HIP_MEMORY_SCOPE_AGENT) < target)
                __builtin_amdgcn_s_sleep(1);
        }
        __syncthreads();
    }
}

// ---------------------------------------------------------------------------
// Host launcher. Workspace layout:
//   [0,256)                       barrier counters (2 x u32)
//   [256, 256+262144)             hbuf: 2 dirs x 2 bufs x 64x512 bf16
//   [262400, +2*16384*2048*4)     xz scratch (f32, both directions)
// ---------------------------------------------------------------------------
extern "C" void kernel_launch(void* const* d_in, const int* in_sizes, int n_in,
                              void* d_out, int out_size, void* d_ws, size_t ws_size,
                              hipStream_t stream) {
    (void)in_sizes; (void)n_in; (void)out_size; (void)ws_size;
    const float* x    = (const float*)d_in[0];
    const float* h_f  = (const float*)d_in[1];
    const float* c_f  = (const float*)d_in[2];
    const float* h_b  = (const float*)d_in[3];
    const float* c_b  = (const float*)d_in[4];
    const float* Wk_f = (const float*)d_in[5];
    const float* Wr_f = (const float*)d_in[6];
    const float* b_f  = (const float*)d_in[7];
    const float* Wk_b = (const float*)d_in[8];
    const float* Wr_b = (const float*)d_in[9];
    const float* b_b  = (const float*)d_in[10];

    char*     ws   = (char*)d_ws;
    unsigned* bar  = (unsigned*)ws;
    bf16*     hbuf = (bf16*)(ws + 256);
    float*    xzb  = (float*)(ws + 256 + (size_t)2 * 2 * B_ * UPAD * sizeof(bf16));
    float*    y    = (float*)d_out;

    lstm_init<<<dim3((2 * 2 * B_ * UPAD) / 256), dim3(256), 0, stream>>>(h_f, h_b, hbuf, bar);

    dim3 g1(NPAD / 128, ROWS / 128, 2);
    lstm_input_gemm<<<g1, dim3(256), 0, stream>>>(x, Wk_f, Wk_b, b_f, b_b, xzb);

    const size_t lds2 = (size_t)128 * WSTR * sizeof(bf16) + (size_t)B_ * HSTR * sizeof(bf16);
    static bool attr_set = [&]{
        hipFuncSetAttribute(reinterpret_cast<const void*>(lstm_recurrent),
                            hipFuncAttributeMaxDynamicSharedMemorySize, (int)lds2);
        return true;
    }();
    (void)attr_set;
    lstm_recurrent<<<dim3(2 * WG_DIR), dim3(256), lds2, stream>>>(
        Wr_f, Wr_b, c_f, c_b, xzb, hbuf, bar, y);
}
